// RealGraphAttentionNetwork_37469294690408
// MI455X (gfx1250) — matrix operand order repaired and checked by reading, not compiled, not run on hardware
//
#include <hip/hip_runtime.h>
#include <hip/hip_bf16.h>
#include <stdint.h>

#define NN   100000
#define EE   1600000
#define ETOT (EE + NN)

typedef __attribute__((ext_vector_type(16))) __bf16   v16bf;
typedef __attribute__((ext_vector_type(8)))  float    v8f;
typedef __attribute__((ext_vector_type(8)))  unsigned v8u;

// ---------------- helpers ----------------
__device__ __forceinline__ unsigned short f2bf(float f) {   // RNE (one-time swizzle only)
  unsigned u = __float_as_uint(f);
  u += 0x7FFFu + ((u >> 16) & 1u);
  return (unsigned short)(u >> 16);
}
// fast truncating pack: one v_perm_b32 packs two f32 high halves -> 2x bf16
__device__ __forceinline__ unsigned packbf(float lo, float hi) {
  return __builtin_amdgcn_perm(__float_as_uint(hi), __float_as_uint(lo), 0x07060302u);
}
// monotone float->uint encoding for atomicMax-based segment max
__device__ __forceinline__ unsigned encf(float f) {
  unsigned u = __float_as_uint(f);
  return (u & 0x80000000u) ? ~u : (u | 0x80000000u);
}
#define ENC_NEG_INF 0x007FFFFFu   // encf(-inf)

// ---------------- weight swizzle: fp32 [K,Nc] -> bf16 B-fragments ----------------
// tile (kt,nt) of 32(K)x16(N); lane<16: n=lane, K=kt*32+0..15 ; lane>=16: n=lane-16, K=kt*32+16..31
__global__ void swizzle_w_kernel(const float* __restrict__ W, unsigned short* __restrict__ out,
                                 int K, int Nc) {
  int tid = blockIdx.x * blockDim.x + threadIdx.x;
  int ntilesN = Nc >> 4;
  int total = (K >> 5) * ntilesN * 32;
  if (tid >= total) return;
  int lane = tid & 31;
  int tile = tid >> 5;
  int kt = tile / ntilesN;
  int nt = tile % ntilesN;
  int n   = nt * 16 + (lane & 15);
  int klo = kt * 32 + ((lane >> 4) << 4);
  unsigned short* o = out + (size_t)tile * 512 + (size_t)lane * 16;
#pragma unroll
  for (int e = 0; e < 16; ++e)
    o[e] = f2bf(W[(size_t)(klo + e) * Nc + n]);
}

// ---------------- GEMM: C[M,Nc] = A[M,K](f32->bf16) x Bswz, WMMA bf16 ----------------
// One wave computes a 16 x (16*NT) strip of C.  KTILES/NTILESN compile-time so every
// global access is base-pointer + immediate offset; explicit register double-buffering
// so iteration kt+1's loads are in flight while kt's WMMAs execute.
template<int KTILES, int NTILESN, int NT>
__global__ void gemm_wmma_kernel(const float* __restrict__ A,
                                 const unsigned short* __restrict__ Bswz,
                                 float* __restrict__ C, int M) {
  constexpr int K  = KTILES * 32;
  constexpr int Nc = NTILESN * 16;
  int wid  = (blockIdx.x * blockDim.x + threadIdx.x) >> 5;
  int lane = threadIdx.x & 31;
  int mtiles  = M >> 4;
  int ngroups = Nc / (16 * NT);
  if (wid >= mtiles * ngroups) return;
  int mt = wid % mtiles;
  int ng = wid / mtiles;

  // A fragment, this code's contiguous k order (same as Bswz): lane%16 = row; lane<16 -> K {0..15}, else {16..31}
  int row   = mt * 16 + (lane & 15);
  int kbase = (lane >> 4) << 4;
  const float* ap = A + (size_t)row * K + kbase;               // all A loads: ap + imm
  const char*  bb = (const char*)(Bswz + ((size_t)(ng * NT) * 32 + lane) * 16); // B: bb + imm

  v8f acc[NT];
#pragma unroll
  for (int t = 0; t < NT; ++t) { v8f z = {}; acc[t] = z; }

  // prologue: stage kt = 0
  float4 a0 = *(const float4*)(ap + 0);
  float4 a1 = *(const float4*)(ap + 4);
  float4 a2 = *(const float4*)(ap + 8);
  float4 a3 = *(const float4*)(ap + 12);
  v8u bcur[NT];
#pragma unroll
  for (int t = 0; t < NT; ++t)
    bcur[t] = *(const v8u*)(bb + t * 1024);

#pragma unroll
  for (int kt = 0; kt < KTILES; ++kt) {
    float4 n0, n1, n2, n3;
    v8u bnxt[NT];
    if (kt + 1 < KTILES) {                      // issue next-stage loads first
      n0 = *(const float4*)(ap + (kt + 1) * 32 + 0);
      n1 = *(const float4*)(ap + (kt + 1) * 32 + 4);
      n2 = *(const float4*)(ap + (kt + 1) * 32 + 8);
      n3 = *(const float4*)(ap + (kt + 1) * 32 + 12);
#pragma unroll
      for (int t = 0; t < NT; ++t)
        bnxt[t] = *(const v8u*)(bb + ((kt + 1) * NTILESN + t) * 1024);
    }
    // pack current A fragment: 8x v_perm_b32 (co-executes with the WMMA pipe)
    v8u au;
    au[0] = packbf(a0.x, a0.y); au[1] = packbf(a0.z, a0.w);
    au[2] = packbf(a1.x, a1.y); au[3] = packbf(a1.z, a1.w);
    au[4] = packbf(a2.x, a2.y); au[5] = packbf(a2.z, a2.w);
    au[6] = packbf(a3.x, a3.y); au[7] = packbf(a3.z, a3.w);
    v16bf av = __builtin_bit_cast(v16bf, au);
#pragma unroll
    for (int t = 0; t < NT; ++t) {
      v16bf bv = __builtin_bit_cast(v16bf, bcur[t]);
      acc[t] = __builtin_amdgcn_wmma_f32_16x16x32_bf16(
          false, av, false, bv, (short)0, acc[t], false, false);
    }
    if (kt + 1 < KTILES) {
      a0 = n0; a1 = n1; a2 = n2; a3 = n3;
#pragma unroll
      for (int t = 0; t < NT; ++t) bcur[t] = bnxt[t];
    }
  }
  // C/D layout: VGPR r -> row r (lanes 0-15) / row 8+r (lanes 16-31); col = lane%16
  int rbase = mt * 16 + ((lane >> 4) << 3);
  int colb  = lane & 15;
#pragma unroll
  for (int t = 0; t < NT; ++t) {
    int col = (ng * NT + t) * 16 + colb;
#pragma unroll
    for (int r = 0; r < 8; ++r)
      C[(size_t)(rbase + r) * Nc + col] = acc[t][r];
  }
}

// ---------------- per-node attention logits alpha_s/alpha_d (wave per node) ----------------
template<int H>
__global__ void alpha_kernel(const float* __restrict__ h, const float* __restrict__ aw,
                             const float* __restrict__ dw, float* __restrict__ as,
                             float* __restrict__ ad, int nnodes) {
  int wid  = (blockIdx.x * blockDim.x + threadIdx.x) >> 5;
  int lane = threadIdx.x & 31;
  if (wid >= nnodes) return;
  const float* hp = h + (size_t)wid * (H * 32);
#pragma unroll
  for (int j = 0; j < H; ++j) {
    float v = hp[j * 32 + lane];                 // coalesced
    float s = v * aw[j * 32 + lane];
    float d = v * dw[j * 32 + lane];
#pragma unroll
    for (int off = 16; off; off >>= 1) {
      s += __shfl_xor(s, off, 32);
      d += __shfl_xor(d, off, 32);
    }
    if (lane == 0) {
      as[(size_t)wid * H + j] = s;
      ad[(size_t)wid * H + j] = d;
    }
  }
}

__global__ void init_emax_kernel(unsigned* __restrict__ e, int n) {
  int i = blockIdx.x * blockDim.x + threadIdx.x;
  if (i < n) e[i] = ENC_NEG_INF;
}
__global__ void fix_emax_kernel(unsigned* __restrict__ e, int n) {
  int i = blockIdx.x * blockDim.x + threadIdx.x;
  if (i >= n) return;
  unsigned u = e[i];
  unsigned bits = (u & 0x80000000u) ? (u & 0x7FFFFFFFu) : ~u;
  float f = __uint_as_float(bits);
  if (!isfinite(f)) f = 0.f;
  ((float*)e)[i] = f;
}

// ---------------- edge pass 1: segment-max of leaky-relu logits ----------------
template<int H>
__global__ void edge_max_kernel(const int* __restrict__ ei, const float* __restrict__ as,
                                const float* __restrict__ ad, unsigned* __restrict__ emax,
                                int nedges, int etot) {
  long long gid = (long long)blockIdx.x * blockDim.x + threadIdx.x;
  if (gid >= (long long)etot * H) return;
  int e  = (int)(gid / H);
  int hh = (int)(gid % H);
  int s, d;
  if (e < nedges) { s = ei[e]; d = ei[nedges + e]; } else { s = d = e - nedges; }
  float v = as[(size_t)s * H + hh] + ad[(size_t)d * H + hh];
  v = v > 0.f ? v : 0.2f * v;
  atomicMax(&emax[(size_t)d * H + hh], encf(v));
}

// ---------------- edge pass 2: exp + denom ----------------
template<int H>
__global__ void edge_exp_kernel(const int* __restrict__ ei, const float* __restrict__ as,
                                const float* __restrict__ ad, const float* __restrict__ emaxf,
                                float* __restrict__ exbuf, float* __restrict__ denom,
                                int nedges, int etot) {
  long long gid = (long long)blockIdx.x * blockDim.x + threadIdx.x;
  if (gid >= (long long)etot * H) return;
  int e  = (int)(gid / H);
  int hh = (int)(gid % H);
  int s, d;
  if (e < nedges) { s = ei[e]; d = ei[nedges + e]; } else { s = d = e - nedges; }
  float v = as[(size_t)s * H + hh] + ad[(size_t)d * H + hh];
  v = v > 0.f ? v : 0.2f * v;
  float ex = __expf(v - emaxf[(size_t)d * H + hh]);
  exbuf[(size_t)e * H + hh] = ex;
  atomicAdd(&denom[(size_t)d * H + hh], ex);
}

// ---------------- edge pass 3: weighted message aggregation ----------------
// wave processes EPW edges; prefetch next edge's h[src] row (global_prefetch_b8)
#define EPW 4
template<int H>
__global__ void edge_agg_kernel(const int* __restrict__ ei, const float* __restrict__ h,
                                const float* __restrict__ exbuf, const float* __restrict__ denom,
                                float* __restrict__ out, int nedges, int etot) {
  int wid  = (blockIdx.x * blockDim.x + threadIdx.x) >> 5;
  int lane = threadIdx.x & 31;
  int ebase = wid * EPW;
  if (ebase >= etot) return;

  int s0;
  if (ebase < nedges) s0 = ei[ebase]; else s0 = ebase - nedges;

#pragma unroll
  for (int k = 0; k < EPW; ++k) {
    int e = ebase + k;
    if (e >= etot) break;
    int s = s0, d;
    if (e < nedges) d = ei[nedges + e]; else d = e - nedges;
    // prefetch next edge's source feature row while we process this one
    if (k + 1 < EPW && e + 1 < etot) {
      int en = e + 1;
      s0 = (en < nedges) ? ei[en] : en - nedges;
      __builtin_prefetch(h + (size_t)s0 * (H * 32) + lane, 0, 1);
    }
    const float* hp = h + (size_t)s * (H * 32);
    float* op = out + (size_t)d * (H * 32);
#pragma unroll
    for (int j = 0; j < H; ++j) {
      float w = exbuf[(size_t)e * H + j] / (denom[(size_t)d * H + j] + 1e-16f);
      atomicAdd(&op[j * 32 + lane], hp[j * 32 + lane] * w);   // coalesced lanes
    }
  }
}

// ---------------- epilogues ----------------
__global__ void bias_relu_kernel(float* __restrict__ x, const float* __restrict__ b,
                                 long long total, int dim) {
  long long i = (long long)blockIdx.x * blockDim.x + threadIdx.x;
  if (i >= total) return;
  float v = x[i] + b[(int)(i % dim)];
  x[i] = v > 0.f ? v : 0.f;
}
__global__ void bias_relu_add_kernel(float* __restrict__ x, const float* __restrict__ b,
                                     const float* __restrict__ res, long long total, int dim) {
  long long i = (long long)blockIdx.x * blockDim.x + threadIdx.x;
  if (i >= total) return;
  float v = x[i] + b[(int)(i % dim)];
  x[i] = (v > 0.f ? v : 0.f) + res[i];
}

// ---------------- log-softmax over 32 channels (wave per node, wave32) ----------------
__global__ void logsoftmax_kernel(const float* __restrict__ agg, const float* __restrict__ b,
                                  float* __restrict__ out, int nnodes) {
  int wid  = (blockIdx.x * blockDim.x + threadIdx.x) >> 5;
  int lane = threadIdx.x & 31;
  if (wid >= nnodes) return;
  float v = agg[(size_t)wid * 32 + lane] + b[lane];
  float m = v;
#pragma unroll
  for (int off = 16; off; off >>= 1) m = fmaxf(m, __shfl_xor(m, off, 32));
  float ex = __expf(v - m);
  float ssum = ex;
#pragma unroll
  for (int off = 16; off; off >>= 1) ssum += __shfl_xor(ssum, off, 32);
  out[(size_t)wid * 32 + lane] = v - m - __logf(ssum);
}

// ---------------- host ----------------
extern "C" void kernel_launch(void* const* d_in, const int* in_sizes, int n_in,
                              void* d_out, int out_size, void* d_ws, size_t ws_size,
                              hipStream_t stream) {
  (void)in_sizes; (void)n_in; (void)out_size; (void)ws_size;
  const float* x   = (const float*)d_in[0];
  const int*   ei  = (const int*)d_in[1];
  const float* W1  = (const float*)d_in[2];
  const float* aS1 = (const float*)d_in[3];
  const float* aD1 = (const float*)d_in[4];
  const float* b1  = (const float*)d_in[5];
  const float* W2  = (const float*)d_in[6];
  const float* aS2 = (const float*)d_in[7];
  const float* aD2 = (const float*)d_in[8];
  const float* b2  = (const float*)d_in[9];
  const float* W3  = (const float*)d_in[10];
  const float* aS3 = (const float*)d_in[11];
  const float* aD3 = (const float*)d_in[12];
  const float* b3  = (const float*)d_in[13];
  float* out = (float*)d_out;

  char* ws = (char*)d_ws;
  size_t off = 0;
  auto alloc = [&](size_t bytes) -> char* {
    char* p = ws + off;
    off = (off + bytes + 255) & ~(size_t)255;
    return p;
  };
  float*          hbuf  = (float*)alloc((size_t)NN * 256 * 4);  // h1 / h2 / h3
  float*          act1  = (float*)alloc((size_t)NN * 256 * 4);  // agg1 -> act1 (residual)
  float*          act2  = (float*)alloc((size_t)NN * 256 * 4);  // agg2 -> act2
  float*          agg3  = (float*)alloc((size_t)NN * 32 * 4);
  float*          asb   = (float*)alloc((size_t)NN * 8 * 4);
  float*          adb   = (float*)alloc((size_t)NN * 8 * 4);
  unsigned*       emax  = (unsigned*)alloc((size_t)NN * 8 * 4); // reused as float after fix
  float*          denom = (float*)alloc((size_t)NN * 8 * 4);
  float*          exbuf = (float*)alloc((size_t)ETOT * 8 * 4);
  unsigned short* w1s   = (unsigned short*)alloc((size_t)128 * 256 * 2);
  unsigned short* w2s   = (unsigned short*)alloc((size_t)256 * 256 * 2);
  unsigned short* w3s   = (unsigned short*)alloc((size_t)256 * 32 * 2);

  // one-time weight swizzle (tiny)
  swizzle_w_kernel<<<(4 * 16 * 32 + 255) / 256, 256, 0, stream>>>(W1, w1s, 128, 256);
  swizzle_w_kernel<<<(8 * 16 * 32 + 255) / 256, 256, 0, stream>>>(W2, w2s, 256, 256);
  swizzle_w_kernel<<<(8 * 2  * 32 + 255) / 256, 256, 0, stream>>>(W3, w3s, 256, 32);

  const int aggwaves = (ETOT + EPW - 1) / EPW;

  auto run_edges8 = [&](const float* aW, const float* dW, float* aggb) {
    alpha_kernel<8><<<(NN * 32 + 255) / 256, 256, 0, stream>>>(hbuf, aW, dW, asb, adb, NN);
    init_emax_kernel<<<(NN * 8 + 255) / 256, 256, 0, stream>>>(emax, NN * 8);
    hipMemsetAsync(denom, 0, (size_t)NN * 8 * 4, stream);
    hipMemsetAsync(aggb, 0, (size_t)NN * 256 * 4, stream);
    edge_max_kernel<8><<<(ETOT * 8 + 255) / 256, 256, 0, stream>>>(ei, asb, adb, emax, EE, ETOT);
    fix_emax_kernel<<<(NN * 8 + 255) / 256, 256, 0, stream>>>(emax, NN * 8);
    edge_exp_kernel<8><<<(ETOT * 8 + 255) / 256, 256, 0, stream>>>(ei, asb, adb, (const float*)emax,
                                                                   exbuf, denom, EE, ETOT);
    edge_agg_kernel<8><<<((long long)aggwaves * 32 + 255) / 256, 256, 0, stream>>>(
        ei, hbuf, exbuf, denom, aggb, EE, ETOT);
  };

  // ---- layer 1 : h1 = x @ W1 (K=128, Nc=256) ----
  {
    int waves = (NN / 16) * (256 / 64);
    gemm_wmma_kernel<4, 16, 4><<<(waves * 32 + 255) / 256, 256, 0, stream>>>(x, w1s, hbuf, NN);
  }
  run_edges8(aS1, aD1, act1);
  bias_relu_kernel<<<((long long)NN * 256 + 255) / 256, 256, 0, stream>>>(act1, b1,
                                                                          (long long)NN * 256, 256);

  // ---- layer 2 (+residual) : h2 = act1 @ W2 (K=256, Nc=256) ----
  {
    int waves = (NN / 16) * (256 / 64);
    gemm_wmma_kernel<8, 16, 4><<<(waves * 32 + 255) / 256, 256, 0, stream>>>(act1, w2s, hbuf, NN);
  }
  run_edges8(aS2, aD2, act2);
  bias_relu_add_kernel<<<((long long)NN * 256 + 255) / 256, 256, 0, stream>>>(act2, b2, act1,
                                                                              (long long)NN * 256, 256);

  // ---- layer 3 (H=1, C=32) : h3 = act2 @ W3 (K=256, Nc=32) ----
  {
    int waves = (NN / 16) * (32 / 32);
    gemm_wmma_kernel<8, 2, 2><<<(waves * 32 + 255) / 256, 256, 0, stream>>>(act2, w3s, hbuf, NN);
    alpha_kernel<1><<<(NN * 32 + 255) / 256, 256, 0, stream>>>(hbuf, aS3, aD3, asb, adb, NN);
    init_emax_kernel<<<(NN + 255) / 256, 256, 0, stream>>>(emax, NN);
    hipMemsetAsync(denom, 0, (size_t)NN * 4, stream);
    hipMemsetAsync(agg3, 0, (size_t)NN * 32 * 4, stream);
    edge_max_kernel<1><<<(ETOT + 255) / 256, 256, 0, stream>>>(ei, asb, adb, emax, EE, ETOT);
    fix_emax_kernel<<<(NN + 255) / 256, 256, 0, stream>>>(emax, NN);
    edge_exp_kernel<1><<<(ETOT + 255) / 256, 256, 0, stream>>>(ei, asb, adb, (const float*)emax,
                                                               exbuf, denom, EE, ETOT);
    edge_agg_kernel<1><<<((long long)aggwaves * 32 + 255) / 256, 256, 0, stream>>>(
        ei, hbuf, exbuf, denom, agg3, EE, ETOT);
  }

  // ---- log-softmax ----
  logsoftmax_kernel<<<((long long)NN * 32 + 255) / 256, 256, 0, stream>>>(agg3, b3, out, NN);
}
